// RFNet_38482906972459
// MI455X (gfx1250) — compile-verified, matrix operand order checked by reading, not run on hardware
//
#include <hip/hip_runtime.h>
#include <hip/hip_bf16.h>

// ---------------------------------------------------------------------------
// Types for CDNA5 WMMA (wave32): 16x16x32 bf16 -> f32 accumulate
// ---------------------------------------------------------------------------
typedef __attribute__((ext_vector_type(16))) __bf16 v16bf;
typedef __attribute__((ext_vector_type(8)))  float  v8f;

namespace {
constexpr int kB    = 32;
constexpr int kNO   = 100;
constexpr int kNE   = 600;
constexpr int kNK   = 1000;
constexpr int kKE   = 1000;
constexpr int kR    = 8;
constexpr int kH    = 512;
constexpr int kIMG  = 2048;
constexpr int kEMB  = 300;
constexpr int kEMBP = 320;   // EMB padded to multiple of 32 for WMMA K loop
constexpr int kC    = 3000;
constexpr int kPAD  = 1;
constexpr int LP    = 40;    // LDS pitch in bf16 (80B): 16B-aligned frag chunks,
                             // conflict-free ds_load_b128 across 16 lanes
}

// ---------------------------------------------------------------------------
// Software-pipelined WMMA GEMM:  C[M,N] = act(A[M,K] @ B[K,N] + bias)
// fp32 in memory, bf16 in LDS/registers, f32 accumulate.
// 128x64 tile, 8 waves (256 threads), K stepped by 32, LDS double-buffered:
//   iter s:  issue global loads for tile s+1 (registers, no wait)
//            -> WMMAs from LDS[cur]  (loads in flight behind them)
//            -> cvt+store regs to LDS[nxt]
//            -> one barrier
// Requires: M%128==0, N%64==0, K%32==0 (all call sites satisfy this).
// ---------------------------------------------------------------------------
template <bool TANH>
__global__ __launch_bounds__(256)
void gemm_wmma_bf16(const float* __restrict__ A, const float* __restrict__ Bm,
                    const float* __restrict__ bias, float* __restrict__ Cm,
                    int M, int N, int K)
{
    __shared__ __bf16 As[2][128 * LP];      // A tile: [row][k]
    __shared__ __bf16 Bs[2][64 * LP];       // B tile transposed: [n][k]

    const int t  = threadIdx.x;
    const int w  = t >> 5;                  // wave id 0..7 -> 16-row block
    const int l  = t & 31;
    const int lr = l & 15;                  // lane row/col within 16
    const int lh = l >> 4;                  // lane half (K-interleave select)
    const int m0 = blockIdx.y * 128;
    const int n0 = blockIdx.x * 64;

    v8f acc[4] = {};
    float4 ar[4], br[2];                    // register staging (pipeline)

    auto loadTiles = [&](int k0) {          // issue 6 global_load_b128, no wait
#pragma unroll
        for (int j = 0; j < 4; ++j) {
            int idx = (t + 256 * j) * 4;    // A: 128x32 = 1024 float4
            int r = idx >> 5, c = idx & 31;
            ar[j] = *(const float4*)(A + (size_t)(m0 + r) * K + k0 + c);
        }
#pragma unroll
        for (int j = 0; j < 2; ++j) {
            int idx = (t + 256 * j) * 4;    // B: 32x64 = 512 float4
            int kr = idx >> 6, c = idx & 63;
            br[j] = *(const float4*)(Bm + (size_t)(k0 + kr) * N + n0 + c);
        }
    };
    auto storeTiles = [&](int buf) {        // fp32 -> bf16, write LDS
#pragma unroll
        for (int j = 0; j < 4; ++j) {
            int idx = (t + 256 * j) * 4;
            int r = idx >> 5, c = idx & 31;
            __bf16* dst = &As[buf][r * LP + c];
            dst[0] = (__bf16)ar[j].x; dst[1] = (__bf16)ar[j].y;
            dst[2] = (__bf16)ar[j].z; dst[3] = (__bf16)ar[j].w;
        }
#pragma unroll
        for (int j = 0; j < 2; ++j) {
            int idx = (t + 256 * j) * 4;
            int kr = idx >> 6, c = idx & 63;
            Bs[buf][(c + 0) * LP + kr] = (__bf16)br[j].x;
            Bs[buf][(c + 1) * LP + kr] = (__bf16)br[j].y;
            Bs[buf][(c + 2) * LP + kr] = (__bf16)br[j].z;
            Bs[buf][(c + 3) * LP + kr] = (__bf16)br[j].w;
        }
    };

    loadTiles(0);
    storeTiles(0);
    __syncthreads();

    const int steps = K / 32;
    for (int s = 0; s < steps; ++s) {
        const int  cur  = s & 1;
        const bool more = (s + 1 < steps);
        if (more) loadTiles((s + 1) * 32);  // overlap with WMMAs below

        // A fragment (16x32): lane<16 holds K 0..7 & 16..23; lane>=16 K 8..15 & 24..31
        v16bf af;
        const __bf16* ap = &As[cur][(16 * w + lr) * LP + lh * 8];
#pragma unroll
        for (int j = 0; j < 8; ++j) { af[j] = ap[j]; af[8 + j] = ap[16 + j]; }
#pragma unroll
        for (int c4 = 0; c4 < 4; ++c4) {
            v16bf bf;
            const __bf16* bp = &Bs[cur][(16 * c4 + lr) * LP + lh * 8];
#pragma unroll
            for (int j = 0; j < 8; ++j) { bf[j] = bp[j]; bf[8 + j] = bp[16 + j]; }
            acc[c4] = __builtin_amdgcn_wmma_f32_16x16x32_bf16(
                false, af, false, bf, (short)0, acc[c4], false, false);
        }
        if (more) storeTiles(cur ^ 1);      // waits loads here, after WMMAs issued
        __syncthreads();
    }

    // ---- epilogue: C/D layout lane l -> col l&15, vgpr i -> row i + 8*(l>>4)
#pragma unroll
    for (int c4 = 0; c4 < 4; ++c4) {
        int col = n0 + 16 * c4 + lr;
        float bv = bias ? bias[col] : 0.0f;
#pragma unroll
        for (int i = 0; i < 8; ++i) {
            int row = m0 + 16 * w + i + 8 * lh;
            float vv = acc[c4][i] + bv;
            if (TANH) vv = tanhf(vv);
            Cm[(size_t)row * N + col] = vv;
        }
    }
}

// ---------------------------------------------------------------------------
// Masked mean-pool of word embeddings: out[row, 0..319] (zero padded 300..319)
// ---------------------------------------------------------------------------
__global__ void pool_kernel(const int* __restrict__ toks,
                            const float* __restrict__ embed,
                            float* __restrict__ out, int* __restrict__ mask,
                            int rows)
{
    int row = blockIdx.x;
    if (row >= rows) return;
    const int* tp = toks + (size_t)row * 8;
    int tk[8]; int cnt = 0;
#pragma unroll
    for (int j = 0; j < 8; ++j) { tk[j] = tp[j]; cnt += (tk[j] != kPAD); }
    float inv = 1.0f / (float)(cnt > 0 ? cnt : 1);
    for (int e = threadIdx.x; e < kEMBP; e += blockDim.x) {
        float acc = 0.0f;
        if (e < kEMB) {
#pragma unroll
            for (int j = 0; j < 8; ++j)
                if (tk[j] != kPAD) acc += embed[(size_t)tk[j] * kEMB + e];
            acc *= inv;
        }
        out[(size_t)row * kEMBP + e] = acc;
    }
    if (mask && threadIdx.x == 0) mask[row] = (cnt > 0) ? 1 : 0;
}

// Zero-pad W_lang [300,512] -> [320,512]
__global__ void pad_w_kernel(const float* __restrict__ W, float* __restrict__ Wp)
{
    int idx = blockIdx.x * blockDim.x + threadIdx.x;
    if (idx >= kEMBP * kH) return;
    int k = idx >> 9;
    Wp[idx] = (k < kEMB) ? W[idx] : 0.0f;
}

// img_node_feat = img_f + img_loc @ W_loc + b_loc   (K=5, elementwise-cheap)
__global__ void loc_add_kernel(const float* __restrict__ img_f,
                               const float* __restrict__ img_loc,
                               const float* __restrict__ W_loc,
                               const float* __restrict__ b_loc,
                               float* __restrict__ out, int rows)
{
    int idx = blockIdx.x * blockDim.x + threadIdx.x;
    if (idx >= rows * kH) return;
    int r = idx >> 9, h = idx & 511;
    const float* lp = img_loc + (size_t)r * 5;
    float acc = b_loc[h];
#pragma unroll
    for (int j = 0; j < 5; ++j) acc += lp[j] * W_loc[j * kH + h];
    out[idx] = img_f[idx] + acc;
}

// Aedge[b*NE+e, 0..1023] = [img_f[n1] | img_f[n2]]
__global__ void gather_edge_kernel(const float* __restrict__ img_f,
                                   const int* __restrict__ n1,
                                   const int* __restrict__ n2,
                                   float* __restrict__ Aedge)
{
    const size_t total = (size_t)kB * kNE * (2 * kH);
    for (size_t idx = (size_t)blockIdx.x * blockDim.x + threadIdx.x; idx < total;
         idx += (size_t)gridDim.x * blockDim.x) {
        size_t row = idx >> 10;          // b*NE+e
        int c = (int)(idx & 1023);
        int b = (int)(row / kNE), e = (int)(row % kNE);
        int nid = (c < kH) ? n1[b * kNE + e] : n2[b * kNE + e];
        int h = c & 511;
        Aedge[idx] = img_f[((size_t)b * kNO + nid) * kH + h];
    }
}

// Score for one key segment, root program node only (r = R-1).
// wave-per-row: s[b, segOffset+n] = sum_m v[m]*tanh(qp[b,R-1,m] + kp[b*segRows+n, m])
__global__ __launch_bounds__(256)
void score_kernel(const float* __restrict__ qp, const float* __restrict__ kp,
                  const float* __restrict__ v, const int* __restrict__ mask,
                  float* __restrict__ s, int segRows, int segOffset, int Ntot)
{
    __shared__ float qs[kH];
    __shared__ float vs[kH];
    const int b = blockIdx.y;
    for (int i = threadIdx.x; i < kH; i += blockDim.x) {
        qs[i] = qp[((size_t)b * kR + (kR - 1)) * kH + i];
        vs[i] = v[i];
    }
    __syncthreads();
    const int wave = threadIdx.x >> 5, lane = threadIdx.x & 31;
    const int n = blockIdx.x * 8 + wave;
    if (n >= segRows) return;
    const float* kr = kp + ((size_t)b * segRows + n) * kH;
    float acc = 0.0f;
    for (int m = lane; m < kH; m += 32)
        acc += vs[m] * tanhf(qs[m] + kr[m]);
    for (int off = 16; off; off >>= 1) acc += __shfl_down(acc, off, 32);
    if (lane == 0) {
        float val = acc;
        if (mask && mask[(size_t)b * segRows + n] == 0) val = -1e9f;
        s[(size_t)b * Ntot + segOffset + n] = val;
    }
}

// Softmax over Ntot keys + weighted sum of two key segments -> out[b, 512]
__global__ __launch_bounds__(256)
void attn_out_kernel(const float* __restrict__ s,
                     const float* __restrict__ keys0, int rows0,
                     const float* __restrict__ keys1, int rows1,
                     float* __restrict__ out, int Ntot)
{
    __shared__ float wbuf[1600];
    __shared__ float rbuf[256];
    const int b = blockIdx.x, tid = threadIdx.x;

    float mx = -1e30f;
    for (int n = tid; n < Ntot; n += 256) {
        float v = s[(size_t)b * Ntot + n];
        wbuf[n] = v; mx = fmaxf(mx, v);
    }
    rbuf[tid] = mx; __syncthreads();
    for (int st = 128; st; st >>= 1) { if (tid < st) rbuf[tid] = fmaxf(rbuf[tid], rbuf[tid + st]); __syncthreads(); }
    mx = rbuf[0]; __syncthreads();

    float sum = 0.0f;
    for (int n = tid; n < Ntot; n += 256) { float e = __expf(wbuf[n] - mx); wbuf[n] = e; sum += e; }
    rbuf[tid] = sum; __syncthreads();
    for (int st = 128; st; st >>= 1) { if (tid < st) rbuf[tid] += rbuf[tid + st]; __syncthreads(); }
    const float inv = 1.0f / rbuf[0]; __syncthreads();

    for (int h = tid; h < kH; h += 256) {
        float acc = 0.0f;
        for (int n = 0; n < rows0; ++n) acc += wbuf[n] * keys0[((size_t)b * rows0 + n) * kH + h];
        for (int n = 0; n < rows1; ++n) acc += wbuf[rows0 + n] * keys1[((size_t)b * rows1 + n) * kH + h];
        out[(size_t)b * kH + h] = acc * inv;
    }
}

// Head: final = node_re + edge_ctx; out = tanh(final@W1+b1) @ W2 + b2  (tiny)
__global__ __launch_bounds__(512)
void head_kernel(const float* __restrict__ node_re, const float* __restrict__ edge_ctx,
                 const float* __restrict__ W1, const float* __restrict__ b1,
                 const float* __restrict__ W2, const float* __restrict__ b2,
                 float* __restrict__ out)
{
    const int b = blockIdx.x, tid = threadIdx.x;
    __shared__ float fin[kH];
    __shared__ float h1[kH];
    fin[tid] = node_re[(size_t)b * kH + tid] + edge_ctx[(size_t)b * kH + tid];
    __syncthreads();
    float acc = b1[tid];
    for (int k = 0; k < kH; ++k) acc += fin[k] * W1[(size_t)k * kH + tid];
    h1[tid] = tanhf(acc);
    __syncthreads();
    for (int c = tid; c < kC; c += 512) {
        float a = b2[c];
        for (int k = 0; k < kH; ++k) a += h1[k] * W2[(size_t)k * kC + c];
        out[(size_t)b * kC + c] = a;
    }
}

// ---------------------------------------------------------------------------
extern "C" void kernel_launch(void* const* d_in, const int* in_sizes, int n_in,
                              void* d_out, int out_size, void* d_ws, size_t ws_size,
                              hipStream_t stream)
{
    (void)in_sizes; (void)n_in; (void)out_size; (void)ws_size;

    const float* img_feat = (const float*)d_in[0];
    const float* img_loc  = (const float*)d_in[1];
    const int*   n1       = (const int*)d_in[2];
    const int*   n2       = (const int*)d_in[3];
    const int*   kg_ent   = (const int*)d_in[4];
    const int*   kg_edg_t = (const int*)d_in[7];
    const int*   r_nodes  = (const int*)d_in[8];
    const float* embed    = (const float*)d_in[11];
    const float* W_lang   = (const float*)d_in[12];
    const float* b_lang   = (const float*)d_in[13];
    const float* W_img    = (const float*)d_in[14];
    const float* b_img    = (const float*)d_in[15];
    const float* W_loc    = (const float*)d_in[16];
    const float* b_loc    = (const float*)d_in[17];
    const float* W_rel    = (const float*)d_in[18];
    const float* b_rel    = (const float*)d_in[19];
    const float* Wq_n     = (const float*)d_in[20];
    const float* Wk_n     = (const float*)d_in[21];
    const float* v_n      = (const float*)d_in[22];
    const float* Wq_e     = (const float*)d_in[23];
    const float* Wk_e     = (const float*)d_in[24];
    const float* v_e      = (const float*)d_in[25];
    const float* W1       = (const float*)d_in[26];
    const float* b1       = (const float*)d_in[27];
    const float* W2       = (const float*)d_in[28];
    const float* b2       = (const float*)d_in[29];
    float* out = (float*)d_out;

    // ---- workspace layout (floats) ----
    float* ws = (float*)d_ws;
    size_t off = 0;
    auto alloc = [&](size_t n) { float* p = ws + off; off += (n + 31) & ~(size_t)31; return p; };
    float* Wl       = alloc((size_t)kEMBP * kH);       // padded W_lang
    float* r_pool   = alloc((size_t)kB * kR * kEMBP);
    float* r_feat   = alloc((size_t)kB * kR * kH);
    float* qp_n     = alloc((size_t)kB * kR * kH);
    float* qp_e     = alloc((size_t)kB * kR * kH);
    float* img_f    = alloc((size_t)kB * kNO * kH);
    float* img_node = alloc((size_t)kB * kNO * kH);
    float* img_edge = alloc((size_t)kB * kNE * kH);
    float* kg_node  = alloc((size_t)kB * kNK * kH);
    float* kg_edge  = alloc((size_t)kB * kKE * kH);
    int*   kg_mask  = (int*)alloc((size_t)kB * kNK);
    float* s_node   = alloc((size_t)kB * (kNO + kNK));
    float* s_edge   = alloc((size_t)kB * (kNE + kKE));
    float* node_re  = alloc((size_t)kB * kH);
    float* edge_ctx = alloc((size_t)kB * kH);
    // shared scratch: pooled embeds (10.24M) / edge concat (19.66M) / kp (16.38M)
    float* scratch  = alloc((size_t)kB * kNE * 2 * kH);

    const dim3 gb(256), gb256(256);

    // ---- language encodes (WMMA lang proj, K padded 300->320) ----
    pad_w_kernel<<<dim3((kEMBP * kH + 255) / 256), gb256, 0, stream>>>(W_lang, Wl);

    pool_kernel<<<dim3(kB * kR), dim3(64), 0, stream>>>(r_nodes, embed, r_pool, nullptr, kB * kR);
    gemm_wmma_bf16<true><<<dim3(kH / 64, (kB * kR) / 128), gb, 0, stream>>>(
        r_pool, Wl, b_lang, r_feat, kB * kR, kH, kEMBP);

    pool_kernel<<<dim3(kB * kNK), dim3(64), 0, stream>>>(kg_ent, embed, scratch, kg_mask, kB * kNK);
    gemm_wmma_bf16<true><<<dim3(kH / 64, (kB * kNK) / 128), gb, 0, stream>>>(
        scratch, Wl, b_lang, kg_node, kB * kNK, kH, kEMBP);

    pool_kernel<<<dim3(kB * kKE), dim3(64), 0, stream>>>(kg_edg_t, embed, scratch, nullptr, kB * kKE);
    gemm_wmma_bf16<true><<<dim3(kH / 64, (kB * kKE) / 128), gb, 0, stream>>>(
        scratch, Wl, b_lang, kg_edge, kB * kKE, kH, kEMBP);

    // ---- image graph features ----
    gemm_wmma_bf16<false><<<dim3(kH / 64, (kB * kNO) / 128), gb, 0, stream>>>(
        img_feat, W_img, b_img, img_f, kB * kNO, kH, kIMG);
    loc_add_kernel<<<dim3((kB * kNO * kH + 255) / 256), gb256, 0, stream>>>(
        img_f, img_loc, W_loc, b_loc, img_node, kB * kNO);
    gather_edge_kernel<<<dim3(8192), gb256, 0, stream>>>(img_f, n1, n2, scratch);
    gemm_wmma_bf16<false><<<dim3(kH / 64, (kB * kNE) / 128), gb, 0, stream>>>(
        scratch, W_rel, b_rel, img_edge, kB * kNE, kH, 2 * kH);

    // ---- query projections (all R rows; only r=R-1 consumed downstream) ----
    gemm_wmma_bf16<false><<<dim3(kH / 64, (kB * kR) / 128), gb, 0, stream>>>(
        r_feat, Wq_n, nullptr, qp_n, kB * kR, kH, kH);
    gemm_wmma_bf16<false><<<dim3(kH / 64, (kB * kR) / 128), gb, 0, stream>>>(
        r_feat, Wq_e, nullptr, qp_e, kB * kR, kH, kH);

    // ---- Find over nodes: kp per segment (scratch reused), scores, softmax+wsum ----
    const int Nn = kNO + kNK;
    gemm_wmma_bf16<false><<<dim3(kH / 64, (kB * kNO) / 128), gb, 0, stream>>>(
        img_node, Wk_n, nullptr, scratch, kB * kNO, kH, kH);
    score_kernel<<<dim3((kNO + 7) / 8, kB), gb256, 0, stream>>>(
        qp_n, scratch, v_n, nullptr, s_node, kNO, 0, Nn);
    gemm_wmma_bf16<false><<<dim3(kH / 64, (kB * kNK) / 128), gb, 0, stream>>>(
        kg_node, Wk_n, nullptr, scratch, kB * kNK, kH, kH);
    score_kernel<<<dim3((kNK + 7) / 8, kB), gb256, 0, stream>>>(
        qp_n, scratch, v_n, kg_mask, s_node, kNK, kNO, Nn);
    attn_out_kernel<<<dim3(kB), gb256, 0, stream>>>(
        s_node, img_node, kNO, kg_node, kNK, node_re, Nn);

    // ---- Find over edges ----
    const int Ne = kNE + kKE;
    gemm_wmma_bf16<false><<<dim3(kH / 64, (kB * kNE) / 128), gb, 0, stream>>>(
        img_edge, Wk_e, nullptr, scratch, kB * kNE, kH, kH);
    score_kernel<<<dim3((kNE + 7) / 8, kB), gb256, 0, stream>>>(
        qp_e, scratch, v_e, nullptr, s_edge, kNE, 0, Ne);
    gemm_wmma_bf16<false><<<dim3(kH / 64, (kB * kKE) / 128), gb, 0, stream>>>(
        kg_edge, Wk_e, nullptr, scratch, kB * kKE, kH, kH);
    score_kernel<<<dim3((kKE + 7) / 8, kB), gb256, 0, stream>>>(
        qp_e, scratch, v_e, nullptr, s_edge, kKE, kNE, Ne);
    attn_out_kernel<<<dim3(kB), gb256, 0, stream>>>(
        s_edge, img_edge, kNE, kg_edge, kKE, edge_ctx, Ne);

    // ---- head ----
    head_kernel<<<dim3(kB), dim3(512), 0, stream>>>(
        node_re, edge_ctx, W1, b1, W2, b2, out);
}
